// VectorQuantizer_28269474743018
// MI455X (gfx1250) — compile-verified, matrix operand order checked by reading, not run on hardware
//
#include <hip/hip_runtime.h>

// ---------------------------------------------------------------------------
// VQ forward for MI455X (gfx1250): distance GEMM via v_wmma_f32_16x16x32_bf16
// with hi/lo bf16-split of f32 operands (3 WMMAs per k-step, f32 accumulate),
// double-buffered B staging via GLOBAL_LOAD_ASYNC_TO_LDS_B128 (ASYNCcnt).
// ---------------------------------------------------------------------------

#define N_ROWS   65536        // 64*1024
#define DIM      256
#define NCODES   1024
#define AP       264          // LDS row pitch (elements) for A tiles (+8 pad)
#define BP       264          // LDS row pitch (elements) for B tiles
#define BBUF     (2 * 64 * BP)  // one B double-buffer slot (hi+lo), elements

typedef __attribute__((ext_vector_type(16))) __bf16 v16bf;
typedef __attribute__((ext_vector_type(8)))  float  v8f;
typedef __attribute__((ext_vector_type(4)))  int    v4i;

union Frag {
  v16bf v;
  uint4 q[2];
};

__device__ inline v8f wmma_bf16(v16bf a, v16bf b, v8f c) {
  // (neg_a, A, neg_b, B, c_mod, C, reuse_a, reuse_b)
  return __builtin_amdgcn_wmma_f32_16x16x32_bf16(false, a, false, b,
                                                 (short)0, c, false, false);
}

// ---- CDNA5 async global->LDS copy (ASYNCcnt) with sync fallback -----------
#if __has_builtin(__builtin_amdgcn_global_load_async_to_lds_b128)
typedef __attribute__((address_space(1))) v4i* gptr_b128;  // global v4i*
typedef __attribute__((address_space(3))) v4i* lptr_b128;  // LDS v4i*
__device__ inline void async_b128(const void* g, void* l) {
  __builtin_amdgcn_global_load_async_to_lds_b128(
      (gptr_b128)g, (lptr_b128)l, 0, 0);
}
#else
__device__ inline void async_b128(const void* g, void* l) {
  *(uint4*)l = *(const uint4*)g;   // synchronous fallback
}
#endif

#if __has_builtin(__builtin_amdgcn_s_wait_asynccnt)
#define WAIT_ASYNC0() __builtin_amdgcn_s_wait_asynccnt(0)
#else
#define WAIT_ASYNC0() asm volatile("s_wait_asynccnt 0" ::: "memory")
#endif

__device__ inline void bf16_split(float x, unsigned short& hb, unsigned short& lb) {
  unsigned u = __builtin_bit_cast(unsigned, x);
  unsigned r = u + 0x7FFFu + ((u >> 16) & 1u);          // RNE to bf16
  hb = (unsigned short)(r >> 16);
  float hf = __builtin_bit_cast(float, ((unsigned)hb) << 16);
  float lo = x - hf;                                    // exact residual
  unsigned ul = __builtin_bit_cast(unsigned, lo);
  unsigned rl = ul + 0x7FFFu + ((ul >> 16) & 1u);
  lb = (unsigned short)(rl >> 16);
}

// ---------------------------------------------------------------------------
// Kernel 1: split codebook into bf16 hi/lo, compute ||e||^2, zero loss slot.
// ---------------------------------------------------------------------------
__global__ __launch_bounds__(DIM) void vq_prep_kernel(
    const float* __restrict__ emb,
    unsigned short* __restrict__ ehi,
    unsigned short* __restrict__ elo,
    float* __restrict__ e2,
    float* __restrict__ loss_slot)
{
  __shared__ float red[DIM];
  const int code = blockIdx.x;
  const int t = threadIdx.x;
  const float x = emb[(size_t)code * DIM + t];
  unsigned short hb, lb;
  bf16_split(x, hb, lb);
  ehi[(size_t)code * DIM + t] = hb;
  elo[(size_t)code * DIM + t] = lb;

  red[t] = x * x;
  __syncthreads();
  for (int s = DIM / 2; s > 0; s >>= 1) {
    if (t < s) red[t] += red[t + s];
    __syncthreads();
  }
  if (t == 0) {
    e2[code] = red[0];
    if (code == 0) *loss_slot = 0.0f;   // re-zeroed every launch (graph-safe)
  }
}

// ---------------------------------------------------------------------------
// Kernel 2: distance GEMM + argmin. 256 threads = 8 waves; block owns 128 rows
// (one 16-row WMMA tile per wave). LDS: A(128x256) bf16 hi/lo + double-
// buffered B chunk (64 codes, bf16 hi/lo) staged with async-to-LDS copies.
// ---------------------------------------------------------------------------
__global__ __launch_bounds__(256) void vq_argmin_kernel(
    const float* __restrict__ inp,
    const unsigned short* __restrict__ ehi,
    const unsigned short* __restrict__ elo,
    const float* __restrict__ e2g,
    int* __restrict__ idxb)
{
  extern __shared__ unsigned short smem[];
  unsigned short* Ahi   = smem;                  // 128*AP
  unsigned short* Alo   = Ahi + 128 * AP;        // 128*AP
  unsigned short* Bbase = Alo + 128 * AP;        // 2 buffers x (2*64*BP)
  float*          e2s   = (float*)(Bbase + 2 * BBUF);  // 2 x 64 floats

  const int tid = threadIdx.x;
  const int r0  = blockIdx.x * 128;

  // Stage A rows (f32 -> bf16 hi/lo) into LDS; float4 coalesced loads.
  for (int i = tid; i < (128 * DIM) / 4; i += 256) {
    const int e4  = i << 2;
    const int row = e4 >> 8;
    const int d   = e4 & (DIM - 1);
    const float4 x = *(const float4*)(inp + (size_t)(r0 + row) * DIM + d);
    const int o = row * AP + d;
    unsigned short hb, lb;
    bf16_split(x.x, hb, lb); Ahi[o + 0] = hb; Alo[o + 0] = lb;
    bf16_split(x.y, hb, lb); Ahi[o + 1] = hb; Alo[o + 1] = lb;
    bf16_split(x.z, hb, lb); Ahi[o + 2] = hb; Alo[o + 2] = lb;
    bf16_split(x.w, hb, lb); Ahi[o + 3] = hb; Alo[o + 3] = lb;
  }

  const int wave = tid >> 5;
  const int lane = tid & 31;
  const int m    = lane & 15;   // row (A) / column (B,C) within 16-tile
  const int h    = lane >> 4;   // half-wave selector in fragment k-layout
  const unsigned aBase = (unsigned)((wave * 16 + m) * AP + h * 8);

  float minv[8];
  int   mini[8];
#pragma unroll
  for (int j = 0; j < 8; ++j) { minv[j] = 3.402823466e38f; mini[j] = 0; }

  // --- B chunk staging (async global -> LDS, double buffered) ---
  auto stageB = [&](int chunk, int buf) {
    unsigned short* Bhi = Bbase + buf * BBUF;
    unsigned short* Blo = Bhi + 64 * BP;
    const unsigned short* sh = ehi + (size_t)chunk * 64 * DIM;
    const unsigned short* sl = elo + (size_t)chunk * 64 * DIM;
    for (int i = tid; i < (64 * DIM) / 8; i += 256) {
      const int code = i >> 5;
      const int d    = (i & 31) << 3;
      async_b128(sh + code * DIM + d, Bhi + code * BP + d);
      async_b128(sl + code * DIM + d, Blo + code * BP + d);
    }
    if (tid < 64) e2s[buf * 64 + tid] = e2g[chunk * 64 + tid];
  };

  stageB(0, 0);   // prologue

  for (int chunk = 0; chunk < NCODES / 64; ++chunk) {
    const int buf = chunk & 1;
    WAIT_ASYNC0();      // my async copies for `chunk` have landed in LDS
    __syncthreads();    // everyone staged `chunk`; everyone done reading buf^1
    if (chunk + 1 < NCODES / 64) stageB(chunk + 1, buf ^ 1);  // overlap w/ math

    const unsigned short* Bhi = Bbase + buf * BBUF;
    const unsigned short* Blo = Bhi + 64 * BP;

#pragma unroll
    for (int ct = 0; ct < 4; ++ct) {
      const unsigned bBase = (unsigned)((ct * 16 + m) * BP + h * 8);
      v8f acc0 = {};   // hi*hi chain
      v8f acc1 = {};   // hi*lo + lo*hi chain (independent -> pipelines)
#pragma unroll
      for (int ks = 0; ks < DIM / 32; ++ks) {
        Frag ah, al, bh, bl;
        const unsigned ao = aBase + ks * 32;
        const unsigned bo = bBase + ks * 32;
        // 16-bit fragment layout: two contiguous 8-element runs per lane
        ah.q[0] = *(const uint4*)(Ahi + ao);  ah.q[1] = *(const uint4*)(Ahi + ao + 16);
        al.q[0] = *(const uint4*)(Alo + ao);  al.q[1] = *(const uint4*)(Alo + ao + 16);
        bh.q[0] = *(const uint4*)(Bhi + bo);  bh.q[1] = *(const uint4*)(Bhi + bo + 16);
        bl.q[0] = *(const uint4*)(Blo + bo);  bl.q[1] = *(const uint4*)(Blo + bo + 16);
        acc0 = wmma_bf16(ah.v, bh.v, acc0);   // hi*hi
        acc1 = wmma_bf16(ah.v, bl.v, acc1);   // hi*lo
        acc1 = wmma_bf16(al.v, bh.v, acc1);   // lo*hi
      }
      const float e2v = e2s[buf * 64 + ct * 16 + m];
      const int   col = chunk * 64 + ct * 16 + m;
#pragma unroll
      for (int j = 0; j < 8; ++j) {
        const float s = acc0[j] + acc1[j];
        const float cand = fmaf(-2.0f, s, e2v);   // ||e||^2 - 2 x.e
        if (cand < minv[j]) { minv[j] = cand; mini[j] = col; }
      }
    }
  }

  // Cross-lane argmin over the 16 lanes that hold each row's 16 columns.
#pragma unroll
  for (int j = 0; j < 8; ++j) {
    float v = minv[j];
    int   ix = mini[j];
    for (int off = 8; off > 0; off >>= 1) {
      const float ov = __shfl_xor(v, off, 16);
      const int   oi = __shfl_xor(ix, off, 16);
      if (ov < v || (ov == v && oi < ix)) { v = ov; ix = oi; }  // first-min tie-break
    }
    mini[j] = ix;
  }
  if (m == 0) {  // lanes 0 (rows 0..7 of tile) and 16 (rows 8..15)
    const int rbase = r0 + wave * 16 + h * 8;
#pragma unroll
    for (int j = 0; j < 8; ++j) idxb[rbase + j] = mini[j];
  }
}

// ---------------------------------------------------------------------------
// Kernel 3: gather quant, write st (== quant), accumulate scaled loss, idx.
// One wave per row; 8 rows per 256-thread block.
// ---------------------------------------------------------------------------
__global__ __launch_bounds__(256) void vq_gather_kernel(
    const float* __restrict__ inp,
    const float* __restrict__ emb,
    const int* __restrict__ idxb,
    float* __restrict__ st,
    float* __restrict__ loss_slot,
    float* __restrict__ oidx)
{
  const int row  = blockIdx.x * 8 + (threadIdx.x >> 5);
  const int lane = threadIdx.x & 31;
  const int k = idxb[row];

  const float* er = emb + (size_t)k * DIM;
  const float* xr = inp + (size_t)row * DIM;
  float*       sr = st  + (size_t)row * DIM;

  float acc = 0.0f;
#pragma unroll
  for (int i = 0; i < 2; ++i) {
    const int d = (lane + i * 32) * 4;
    const float4 e = *(const float4*)(er + d);
    const float4 x = *(const float4*)(xr + d);
    *(float4*)(sr + d) = e;
    const float dx = e.x - x.x, dy = e.y - x.y, dz = e.z - x.z, dw = e.w - x.w;
    acc += dx * dx + dy * dy + dz * dz + dw * dw;
  }
  for (int off = 16; off > 0; off >>= 1) acc += __shfl_xor(acc, off, 32);
  if (lane == 0) {
    // loss = (1 + COMMIT) * mean((quant-inp)^2)
    atomicAdd(loss_slot, acc * (1.25f / ((float)N_ROWS * (float)DIM)));
    oidx[row] = (float)k;
  }
}

// ---------------------------------------------------------------------------
extern "C" void kernel_launch(void* const* d_in, const int* in_sizes, int n_in,
                              void* d_out, int out_size, void* d_ws, size_t ws_size,
                              hipStream_t stream) {
  (void)in_sizes; (void)n_in; (void)out_size; (void)ws_size;
  const float* inp = (const float*)d_in[0];   // [65536, 256] f32
  const float* emb = (const float*)d_in[1];   // [1024, 256]  f32

  // Workspace layout
  unsigned short* ehi = (unsigned short*)d_ws;                 // 512 KB
  unsigned short* elo = ehi + (size_t)NCODES * DIM;            // 512 KB
  float*          e2  = (float*)(elo + (size_t)NCODES * DIM);  // 4 KB
  int*            idxb = (int*)(e2 + NCODES);                  // 256 KB

  // Output layout: [st (N*D)] [loss (1)] [idx (N)]
  float* out  = (float*)d_out;
  float* st   = out;
  float* loss = out + (size_t)N_ROWS * DIM;
  float* oidx = loss + 1;

  vq_prep_kernel<<<NCODES, DIM, 0, stream>>>(emb, ehi, elo, e2, loss);

  const size_t ldsBytes =
      (size_t)(2 * 128 * AP + 2 * BBUF) * sizeof(unsigned short) +
      2 * 64 * sizeof(float);   // ~266 KB (<= 320 KB/WG on gfx1250)
  vq_argmin_kernel<<<N_ROWS / 128, 256, ldsBytes, stream>>>(inp, ehi, elo, e2, idxb);

  vq_gather_kernel<<<N_ROWS / 8, 256, 0, stream>>>(inp, emb, idxb, st, loss, oidx);
}
